// MultiHeadAttention_3272765079878
// MI455X (gfx1250) — matrix operand order repaired and checked by reading, not compiled, not run on hardware
//
#include <hip/hip_runtime.h>
#include <hip/hip_bf16.h>

typedef __bf16 bf16_t;
typedef __attribute__((ext_vector_type(16))) __bf16 v16bf;
typedef __attribute__((ext_vector_type(8)))  float  v8f;
typedef unsigned int u32;
typedef __attribute__((ext_vector_type(4))) u32 tdm_g0_t;
typedef __attribute__((ext_vector_type(8))) int tdm_g1_t;
typedef __attribute__((ext_vector_type(4))) int tdm_g23_t;

#define D_MODEL 1024
#define N_HEADS 16
#define D_HEAD  64
#define SEQ     2048
#define BATCH   2
#define MTOT    (BATCH * SEQ)
#define NEG_INF_F (-1.0e10f)

// ---------------------------------------------------------------------------
// Tensor Data Mover support (CDNA5). 5-arg builtin on ROCm clang-22; 6-arg on
// amdgpu-toolchain (detected via its TDM header). Fallback: cooperative loads.
// ---------------------------------------------------------------------------
#if defined(__gfx1250__) && __has_builtin(__builtin_amdgcn_tensor_load_to_lds) && \
    __has_builtin(__builtin_amdgcn_s_wait_tensorcnt)
#define ATHENA_HAS_TDM 1
#if __has_include(<hip/amd_detail/amd_gfx1250_TDM.h>)
#define ATHENA_TDM_6ARG 1
#else
#define ATHENA_TDM_6ARG 0
#endif
#else
#define ATHENA_HAS_TDM 0
#endif

#if ATHENA_HAS_TDM
// 2-D bf16 tile load: tile_h rows x tile_w elems, row stride in elements.
// D# layout per CDNA5 ISA 8.3/8.4 (Group0: count|lds|global|type, Group1:
// data_size=1 (2B), tensor dims, tile dims, 48-bit dim0 stride).
__device__ __forceinline__ void tdm_load_2d_bf16(u32 lds_off, const bf16_t* gptr,
                                                 u32 tile_w, u32 tile_h,
                                                 u32 tensor_w, u32 tensor_h,
                                                 u32 row_stride) {
  unsigned long long ga = (unsigned long long)gptr;
  tdm_g0_t g0;
  g0[0] = 1u;                                            // count=1, user mode
  g0[1] = lds_off;                                       // lds_addr
  g0[2] = (u32)ga;                                       // global_addr[31:0]
  g0[3] = (u32)((ga >> 32) & 0x01ffffffull) | (2u << 30);// addr[56:32] | type=2
  tdm_g1_t g1;
  g1[0] = (int)(1u << 16);                               // data_size=1 (2 B)
  g1[1] = (int)((tensor_w & 0xffffu) << 16);             // tensor_dim0 lo16
  g1[2] = (int)((tensor_w >> 16) | ((tensor_h & 0xffffu) << 16));
  g1[3] = (int)((tensor_h >> 16) | (tile_w << 16));      // dim1 hi | tile_dim0
  g1[4] = (int)tile_h;                                   // tile_dim1, tile_dim2=0
  g1[5] = (int)row_stride;                               // dim0_stride[31:0]
  g1[6] = 0;                                             // stride hi | dim1_stride
  g1[7] = 0;
  tdm_g23_t gz = {0, 0, 0, 0};
#if ATHENA_TDM_6ARG
  tdm_g1_t gz8 = {0, 0, 0, 0, 0, 0, 0, 0};
  __builtin_amdgcn_tensor_load_to_lds(g0, g1, gz, gz, gz8, 0);
#else
  __builtin_amdgcn_tensor_load_to_lds(g0, g1, gz, gz, 0);
#endif
}
#define TDM_WAIT() __builtin_amdgcn_s_wait_tensorcnt(0)
#endif

// ---------------------------------------------------------------------------
// fp32 -> bf16 conversion (vectorized by 4)
// ---------------------------------------------------------------------------
__global__ void cvt_f32_to_bf16(const float* __restrict__ in,
                                bf16_t* __restrict__ out, int n4) {
  int i = blockIdx.x * blockDim.x + threadIdx.x;
  if (i < n4) {
    float4 f = ((const float4*)in)[i];
    union { bf16_t h[4]; uint2 u; } p;
    p.h[0] = (bf16_t)f.x; p.h[1] = (bf16_t)f.y;
    p.h[2] = (bf16_t)f.z; p.h[3] = (bf16_t)f.w;
    ((uint2*)out)[i] = p.u;
  }
}

// ---------------------------------------------------------------------------
// GEMM: C[M,N] = A[M,K](bf16) @ W[K,N](bf16) + bias(f32).
// 256 threads = 8 waves. Block tile 32(M) x 128(N); wave w owns 16-col strip,
// two M accumulators -> 2 independent WMMA per k-step, B fragment reused.
// Dynamic LDS: As 32x32 bf16 @0 (2048 B, TDM-loadable), Wt 128x32 bf16 @2048
// (transposed so B-frag gathers are contiguous b128, conflict-free).
// ---------------------------------------------------------------------------
template <typename OutT>
__global__ void gemm_bf16_wmma(const bf16_t* __restrict__ A,
                               const bf16_t* __restrict__ W,
                               const float* __restrict__ bias,
                               OutT* __restrict__ C,
                               int M, int N, int K) {
  extern __shared__ char gsm[];
  bf16_t* As = (bf16_t*)gsm;            // 32 x 32
  bf16_t* Wt = (bf16_t*)(gsm + 2048);   // [n 0..127][k 0..31]

  const int tid  = threadIdx.x;
  const int wave = tid >> 5;
  const int lane = tid & 31;
  const int m0 = blockIdx.y * 32;
  const int n0 = blockIdx.x * 128;

  const int rowA = lane & 15;
  const int kbA  = (lane < 16) ? 0 : 8;
  const int colB = lane & 15;
  const int kbB  = (lane < 16) ? 0 : 16;

  v8f acc0 = {}, acc1 = {};
  for (int k0 = 0; k0 < K; k0 += 32) {
    // --- A tile 32x32 ---
#if ATHENA_HAS_TDM
    if (wave == 0)
      tdm_load_2d_bf16(0u, A + (size_t)m0 * K + k0, 32u, 32u,
                       (u32)K, (u32)M, (u32)K);
#else
    {
      int idx = tid * 4;                 // 1024 elems, b64 each
      int r = idx >> 5, c = idx & 31;
      *(uint2*)(&As[idx]) = *(const uint2*)(A + (size_t)(m0 + r) * K + k0 + c);
    }
#endif
    // --- W tile 32x128, stored transposed (manual; TDM cannot transpose) ---
    {
      int r = tid >> 3;                  // k row 0..31
      int cs = (tid & 7) * 16;           // n seg
      const bf16_t* src = W + (size_t)(k0 + r) * N + n0 + cs;
      union { uint4 u4[2]; bf16_t h[16]; } wbuf;
      wbuf.u4[0] = *(const uint4*)(src);
      wbuf.u4[1] = *(const uint4*)(src + 8);
#pragma unroll
      for (int j = 0; j < 16; ++j) Wt[(cs + j) * 32 + r] = wbuf.h[j];
    }
#if ATHENA_HAS_TDM
    if (wave == 0) TDM_WAIT();
#endif
    __syncthreads();

    v16bf a0, a1, b;
#pragma unroll
    for (int i = 0; i < 16; ++i) {
      int k = 2 * kbA + i;
      a0[i] = As[rowA * 32 + k];
      a1[i] = As[(16 + rowA) * 32 + k];
    }
#pragma unroll
    for (int i = 0; i < 16; ++i)
      b[i] = Wt[(wave * 16 + colB) * 32 + kbB + i];

    acc0 = __builtin_amdgcn_wmma_f32_16x16x32_bf16(
        false, a0, false, b, (short)0, acc0, false, false);
    acc1 = __builtin_amdgcn_wmma_f32_16x16x32_bf16(
        false, a1, false, b, (short)0, acc1, false, false);
    __syncthreads();
  }

  const int col = n0 + wave * 16 + colB;
  const float bv = bias[col];
#pragma unroll
  for (int i = 0; i < 8; ++i) {
    int r = i + ((lane < 16) ? 0 : 8);
    C[(size_t)(m0 + r) * N + col]      = (OutT)(acc0[i] + bv);
    C[(size_t)(m0 + 16 + r) * N + col] = (OutT)(acc1[i] + bv);
  }
}

// ---------------------------------------------------------------------------
// Fused attention: one block (128 thr, 4 waves) per (b, h, 16 query rows).
// Dynamic LDS layout (141824 B):
//   sS   @      0 : 16*2048 f32 (scores / probabilities)
//   sQ   @ 131072 : 16*64  bf16
//   sKV  @ 133120 : 4 slots * 16*64 bf16 (K staging; V reuses 32*64)
//   sRed @ 141312 : 16*8 f32
// ---------------------------------------------------------------------------
#define SS_OFF  0u
#define SQ_OFF  131072u
#define SKV_OFF 133120u

__global__ void mha_attn_kernel(const bf16_t* __restrict__ Qp,
                                const bf16_t* __restrict__ Kp,
                                const bf16_t* __restrict__ Vp,
                                const int* __restrict__ mask,
                                float* __restrict__ attn_w,
                                bf16_t* __restrict__ attn_out) {
  extern __shared__ char smem_raw[];
  float*  sS   = (float*)smem_raw;
  bf16_t* sQ   = (bf16_t*)(smem_raw + SQ_OFF);
  bf16_t* sKV  = (bf16_t*)(smem_raw + SKV_OFF);
  float*  sRed = (float*)(smem_raw + 141312);

  const int tid  = threadIdx.x;
  const int wave = tid >> 5;
  const int lane = tid & 31;
  const int q0 = blockIdx.x * 16;
  const int h  = blockIdx.y;
  const int b  = blockIdx.z;
  const size_t seqBase = (size_t)b * SEQ;
  const int hoff = h * D_HEAD;

  const int rowA = lane & 15;
  const int kbA  = (lane < 16) ? 0 : 8;
  const int colB = lane & 15;
  const int kbB  = (lane < 16) ? 0 : 16;

  // ---- Phase 0: stage Q tile (16 x 64 bf16), build A fragments
#if ATHENA_HAS_TDM
  if (wave == 0) {
    tdm_load_2d_bf16(SQ_OFF, Qp + (seqBase + q0) * D_MODEL + hoff,
                     64u, 16u, (u32)D_MODEL, (u32)MTOT, (u32)D_MODEL);
    TDM_WAIT();
  }
#else
  {
    int idx = tid * 8;
    int r = idx >> 6, c = idx & 63;
    *(uint4*)(&sQ[idx]) =
        *(const uint4*)(Qp + (seqBase + q0 + r) * D_MODEL + hoff + c);
  }
#endif
  __syncthreads();

  v16bf aQ0, aQ1;
#pragma unroll
  for (int i = 0; i < 16; ++i) {
    int k = 2 * kbA + i;
    aQ0[i] = sQ[rowA * 64 + k];
    aQ1[i] = sQ[rowA * 64 + 32 + k];
  }
  const float scale = 0.125f;           // 1/sqrt(64)

  // ---- Phase 1: S = scale * Q @ K^T with mask -> LDS (wave-private work,
  //      per-wave K slots + disjoint score columns => no barriers needed)
  for (int iter = 0; iter < 32; ++iter) {
    const int t  = iter * 4 + wave;
    const int n0 = t * 16;
    bf16_t* sK = sKV + wave * (16 * 64);
#if ATHENA_HAS_TDM
    tdm_load_2d_bf16(SKV_OFF + (u32)wave * 2048u,
                     Kp + (seqBase + n0) * D_MODEL + hoff,
                     64u, 16u, (u32)D_MODEL, (u32)MTOT, (u32)D_MODEL);
    TDM_WAIT();
#else
    {
      int r = lane >> 1, half = lane & 1;
      const bf16_t* src = Kp + (seqBase + n0 + r) * D_MODEL + hoff + half * 32;
      uint4* dst = (uint4*)(&sK[r * 64 + half * 32]);
      const uint4* s4 = (const uint4*)src;
      dst[0] = s4[0]; dst[1] = s4[1]; dst[2] = s4[2]; dst[3] = s4[3];
    }
#endif
    v16bf b0, b1;
#pragma unroll
    for (int i = 0; i < 16; ++i) {
      b0[i] = sK[colB * 64 + kbB + i];
      b1[i] = sK[colB * 64 + 32 + kbB + i];
    }
    v8f acc = {};
    acc = __builtin_amdgcn_wmma_f32_16x16x32_bf16(
        false, aQ0, false, b0, (short)0, acc, false, false);
    acc = __builtin_amdgcn_wmma_f32_16x16x32_bf16(
        false, aQ1, false, b1, (short)0, acc, false, false);

#pragma unroll
    for (int i = 0; i < 8; ++i) {
      int r = i + ((lane < 16) ? 0 : 8);
      int n = n0 + colB;
      int m = mask[(seqBase + q0 + r) * SEQ + n];
      sS[r * SEQ + n] = (m == 0) ? NEG_INF_F : acc[i] * scale;
    }
  }
  __syncthreads();

  // ---- Phase 2: softmax over N=2048, 8 threads per row, attn_w streamout
  {
    const int row = tid >> 3;
    const int sub = tid & 7;
    float* rowS = sS + row * SEQ;
    float mx = -3.0e38f;
    for (int n = sub * 256; n < sub * 256 + 256; n += 4) {
      float4 v = *(float4*)(rowS + n);
      mx = fmaxf(mx, fmaxf(fmaxf(v.x, v.y), fmaxf(v.z, v.w)));
    }
    sRed[row * 8 + sub] = mx;
    __syncthreads();
    float rmax = sRed[row * 8];
#pragma unroll
    for (int j = 1; j < 8; ++j) rmax = fmaxf(rmax, sRed[row * 8 + j]);
    __syncthreads();

    float sum = 0.0f;
    for (int n = sub * 256; n < sub * 256 + 256; n += 4) {
      float4 v = *(float4*)(rowS + n);
      v.x = __expf(v.x - rmax); v.y = __expf(v.y - rmax);
      v.z = __expf(v.z - rmax); v.w = __expf(v.w - rmax);
      *(float4*)(rowS + n) = v;
      sum += v.x + v.y + v.z + v.w;
    }
    sRed[row * 8 + sub] = sum;
    __syncthreads();
    float rsum = 0.0f;
#pragma unroll
    for (int j = 0; j < 8; ++j) rsum += sRed[row * 8 + j];
    const float inv = 1.0f / rsum;

    float* gw = attn_w + ((size_t)(b * N_HEADS + h) * SEQ + q0 + row) * SEQ;
    for (int n = sub * 256; n < sub * 256 + 256; n += 4) {
      float4 v = *(float4*)(rowS + n);
      v.x *= inv; v.y *= inv; v.z *= inv; v.w *= inv;
      *(float4*)(rowS + n) = v;
      *(float4*)(gw + n)   = v;
    }
    __syncthreads();
  }

  // ---- Phase 3: O(16x64) = P(16x2048) @ V(2048x64); wave w owns cols w*16..
  v8f accO = {};
  for (int k0 = 0; k0 < SEQ; k0 += 32) {
#if ATHENA_HAS_TDM
    if (wave == 0) {
      tdm_load_2d_bf16(SKV_OFF, Vp + (seqBase + k0) * D_MODEL + hoff,
                       64u, 32u, (u32)D_MODEL, (u32)MTOT, (u32)D_MODEL);
      TDM_WAIT();
    }
#else
    {
      int idx = tid * 16;               // V tile 32x64 row-major
      int r = idx >> 6, c = idx & 63;
      const bf16_t* src = Vp + (seqBase + k0 + r) * D_MODEL + hoff + c;
      *(uint4*)(&sKV[idx])     = *(const uint4*)(src);
      *(uint4*)(&sKV[idx + 8]) = *(const uint4*)(src + 8);
    }
#endif
    __syncthreads();

    v16bf aP, bV;
#pragma unroll
    for (int i = 0; i < 16; ++i) {
      int k = 2 * kbA + i;
      aP[i] = (bf16_t)sS[rowA * SEQ + k0 + k];
      bV[i] = sKV[(kbB + i) * 64 + wave * 16 + colB];
    }
    accO = __builtin_amdgcn_wmma_f32_16x16x32_bf16(
        false, aP, false, bV, (short)0, accO, false, false);
    __syncthreads();
  }

#pragma unroll
  for (int i = 0; i < 8; ++i) {
    int r = i + ((lane < 16) ? 0 : 8);
    attn_out[(seqBase + q0 + r) * D_MODEL + hoff + wave * 16 + colB] =
        (bf16_t)accO[i];
  }
}

// ---------------------------------------------------------------------------
// launcher
// ---------------------------------------------------------------------------
extern "C" void kernel_launch(void* const* d_in, const int* in_sizes, int n_in,
                              void* d_out, int out_size, void* d_ws, size_t ws_size,
                              hipStream_t stream) {
  (void)in_sizes; (void)n_in; (void)out_size; (void)ws_size;
  const float* q    = (const float*)d_in[0];
  const float* k    = (const float*)d_in[1];
  const float* v    = (const float*)d_in[2];
  const int*   mask = (const int*)d_in[3];
  const float* Wq   = (const float*)d_in[4];
  const float* bq   = (const float*)d_in[5];
  const float* Wk   = (const float*)d_in[6];
  const float* bk   = (const float*)d_in[7];
  const float* Wv   = (const float*)d_in[8];
  const float* bv   = (const float*)d_in[9];
  const float* Wo   = (const float*)d_in[10];
  const float* bo   = (const float*)d_in[11];

  float* out    = (float*)d_out;
  float* attn_w = out + (size_t)BATCH * SEQ * D_MODEL;

  char* ws = (char*)d_ws;
  const size_t SZ  = (size_t)MTOT * D_MODEL;
  const size_t SZB = SZ * sizeof(bf16_t);
  bf16_t* q_b  = (bf16_t*)(ws + 0 * SZB);
  bf16_t* k_b  = (bf16_t*)(ws + 1 * SZB);
  bf16_t* v_b  = (bf16_t*)(ws + 2 * SZB);
  bf16_t* Qp   = (bf16_t*)(ws + 3 * SZB);
  bf16_t* Kp   = (bf16_t*)(ws + 4 * SZB);
  bf16_t* Vp   = (bf16_t*)(ws + 5 * SZB);
  bf16_t* ao_b = (bf16_t*)(ws + 6 * SZB);
  const size_t WB = (size_t)D_MODEL * D_MODEL * sizeof(bf16_t);
  bf16_t* Wq_b = (bf16_t*)(ws + 7 * SZB + 0 * WB);
  bf16_t* Wk_b = (bf16_t*)(ws + 7 * SZB + 1 * WB);
  bf16_t* Wv_b = (bf16_t*)(ws + 7 * SZB + 2 * WB);
  bf16_t* Wo_b = (bf16_t*)(ws + 7 * SZB + 3 * WB);

  {
    int n4a = (int)(SZ / 4);
    int n4w = (int)((size_t)D_MODEL * D_MODEL / 4);
    cvt_f32_to_bf16<<<(n4a + 255) / 256, 256, 0, stream>>>(q,  q_b,  n4a);
    cvt_f32_to_bf16<<<(n4a + 255) / 256, 256, 0, stream>>>(k,  k_b,  n4a);
    cvt_f32_to_bf16<<<(n4a + 255) / 256, 256, 0, stream>>>(v,  v_b,  n4a);
    cvt_f32_to_bf16<<<(n4w + 255) / 256, 256, 0, stream>>>(Wq, Wq_b, n4w);
    cvt_f32_to_bf16<<<(n4w + 255) / 256, 256, 0, stream>>>(Wk, Wk_b, n4w);
    cvt_f32_to_bf16<<<(n4w + 255) / 256, 256, 0, stream>>>(Wv, Wv_b, n4w);
    cvt_f32_to_bf16<<<(n4w + 255) / 256, 256, 0, stream>>>(Wo, Wo_b, n4w);
  }

  // QKV projections (bf16 out). Dynamic LDS: 2048 (As) + 8192 (Wt) = 10240 B
  dim3 ggrid(D_MODEL / 128, MTOT / 32);
  const size_t gshm = 10240;
  gemm_bf16_wmma<bf16_t><<<ggrid, 256, gshm, stream>>>(q_b, Wq_b, bq, Qp,
                                                       MTOT, D_MODEL, D_MODEL);
  gemm_bf16_wmma<bf16_t><<<ggrid, 256, gshm, stream>>>(k_b, Wk_b, bk, Kp,
                                                       MTOT, D_MODEL, D_MODEL);
  gemm_bf16_wmma<bf16_t><<<ggrid, 256, gshm, stream>>>(v_b, Wv_b, bv, Vp,
                                                       MTOT, D_MODEL, D_MODEL);

  // fused attention (141824 B dynamic LDS)
  const size_t shmem = 141824;
  hipFuncSetAttribute((const void*)mha_attn_kernel,
                      hipFuncAttributeMaxDynamicSharedMemorySize, (int)shmem);
  mha_attn_kernel<<<dim3(SEQ / 16, N_HEADS, BATCH), 128, shmem, stream>>>(
      Qp, Kp, Vp, mask, attn_w, ao_b);

  // output projection (fp32 out -> d_out)
  gemm_bf16_wmma<float><<<ggrid, 256, gshm, stream>>>(ao_b, Wo_b, bo, out,
                                                      MTOT, D_MODEL, D_MODEL);
}
